// DistFlashAttn_83459804496426
// MI455X (gfx1250) — compile-verified, matrix operand order checked by reading, not run on hardware
//
#include <hip/hip_runtime.h>

typedef __attribute__((ext_vector_type(16))) __bf16 v16bf;
typedef __attribute__((ext_vector_type(8)))  float  v8f;

#define SQ   4096
#define NH   8
#define DH   128
#define BLK  1024
#define QT   64      // q rows per workgroup
#define WQT  16      // q rows per wave
#define KT   32      // kv rows per tile
#define KPAD 136     // K LDS row stride in halfs (16B pad -> conflict-free b128 reads)
#define VPAD 40      // V^T LDS row stride in halfs
#define PPAD 40      // P LDS row stride in halfs

union FragB { v16bf v; unsigned short u[16]; uint4 q[2]; };

// native bf16 convert (backend emits v_cvt_(pk_)bf16_f32 instead of integer rounding)
__device__ __forceinline__ unsigned short f2bf(float f) {
  return __builtin_bit_cast(unsigned short, (__bf16)f);
}
__device__ __forceinline__ unsigned int bfpack(float lo, float hi) {
  return (unsigned int)f2bf(lo) | ((unsigned int)f2bf(hi) << 16);
}

__device__ __forceinline__ float redmax16(float x) {
  x = fmaxf(x, __shfl_xor(x, 1, 32));
  x = fmaxf(x, __shfl_xor(x, 2, 32));
  x = fmaxf(x, __shfl_xor(x, 4, 32));
  x = fmaxf(x, __shfl_xor(x, 8, 32));
  return x;
}
__device__ __forceinline__ float redsum16(float x) {
  x += __shfl_xor(x, 1, 32);
  x += __shfl_xor(x, 2, 32);
  x += __shfl_xor(x, 4, 32);
  x += __shfl_xor(x, 8, 32);
  return x;
}

__global__ __launch_bounds__(128)
void dist_flash_attn_kernel(const float* __restrict__ q,
                            const float* __restrict__ k_local,
                            const float* __restrict__ v_local,
                            const float* __restrict__ k_remote,
                            const float* __restrict__ v_remote,
                            float* __restrict__ out,
                            float* __restrict__ lse) {
  __shared__ unsigned short Klds[KT * KPAD];        // K tile, bf16, [kv][d]
  __shared__ unsigned short Vtld[DH * VPAD];        // V tile transposed, bf16, [d][kv]
  __shared__ unsigned short Plds[4][WQT * PPAD];    // per-wave P tile, bf16

  const int tid  = threadIdx.x;
  const int wave = tid >> 5;
  const int lane = tid & 31;
  const int ln16 = lane & 15;
  const int hf   = lane >> 4;                       // lane half (0/1)
  const int h    = blockIdx.y;
  const int q0wg = blockIdx.x * QT;
  const int q0   = q0wg + wave * WQT;
  const int ks   = (q0wg / BLK) * BLK;              // ranges are identity blocks

  const float scale = 0.08838834764831845f;         // 1/sqrt(128)

  // ---- Q A-fragments: lane row = q0 + ln16; K pattern = {hf*8+0..7, 16+hf*8+0..7} per 32-chunk
  FragB aQ[4];
  {
    const float* qrow = q + ((size_t)(q0 + ln16) * NH + h) * DH;
    #pragma unroll
    for (int kc = 0; kc < 4; ++kc) {
      const int kb = kc * 32 + hf * 8;
      #pragma unroll
      for (int e = 0; e < 8; ++e) {
        aQ[kc].u[e]     = f2bf(qrow[kb + e]);
        aQ[kc].u[8 + e] = f2bf(qrow[kb + 16 + e]);
      }
    }
  }

  v8f o[8];
  const v8f vzero = {0.f,0.f,0.f,0.f,0.f,0.f,0.f,0.f};
  #pragma unroll
  for (int t = 0; t < 8; ++t) o[t] = vzero;
  float m[8], l[8];
  #pragma unroll
  for (int g = 0; g < 8; ++g) { m[g] = -1e30f; l[g] = 0.0f; }

  // cooperative-load index split (K: 4 threads/row; V: 2 rows/thread for packed stores)
  const int kr_  = tid >> 2;          // K: kv row 0..31
  const int kcb  = (tid & 3) * 32;    // K: d col base
  const int vr2  = tid >> 3;          // V: row pair 0..15
  const int vcb  = (tid & 7) * 16;    // V: d col base

  for (int phase = 0; phase < 2; ++phase) {
    const float* Kg = phase ? k_remote : k_local;
    const float* Vg = phase ? v_remote : v_local;
    const bool causal = (phase == 0);
    // remote contributions count twice in the reference merge: score += ln2
    const float bias = phase ? 0.6931471805599453f : 0.0f;
    const int ntiles = phase ? (BLK / KT) : ((q0wg - ks + QT) / KT);

    for (int it = 0; it < ntiles; ++it) {
      const int kv0 = ks + it * KT;

      // ---- cooperative K/V tile load, f32 -> bf16 (V stored transposed, packed pairs) ----
      {
        const float* kr = Kg + ((size_t)(kv0 + kr_) * NH + h) * DH + kcb;
        #pragma unroll
        for (int j = 0; j < 32; j += 4) {
          float4 k4 = *(const float4*)(kr + j);
          *(unsigned int*)&Klds[kr_ * KPAD + kcb + j]     = bfpack(k4.x, k4.y);
          *(unsigned int*)&Klds[kr_ * KPAD + kcb + j + 2] = bfpack(k4.z, k4.w);
        }
        const float* vr0 = Vg + ((size_t)(kv0 + 2 * vr2) * NH + h) * DH + vcb;
        const float* vr1 = vr0 + (size_t)NH * DH;
        #pragma unroll
        for (int j = 0; j < 16; j += 4) {
          float4 a4 = *(const float4*)(vr0 + j);
          float4 b4 = *(const float4*)(vr1 + j);
          *(unsigned int*)&Vtld[(vcb + j + 0) * VPAD + 2 * vr2] = bfpack(a4.x, b4.x);
          *(unsigned int*)&Vtld[(vcb + j + 1) * VPAD + 2 * vr2] = bfpack(a4.y, b4.y);
          *(unsigned int*)&Vtld[(vcb + j + 2) * VPAD + 2 * vr2] = bfpack(a4.z, b4.z);
          *(unsigned int*)&Vtld[(vcb + j + 3) * VPAD + 2 * vr2] = bfpack(a4.w, b4.w);
        }
      }
      // prefetch next tile while this one is consumed (global_prefetch_b8)
      if (it + 1 < ntiles) {
        __builtin_prefetch(Kg + ((size_t)(kv0 + KT + kr_) * NH + h) * DH + kcb, 0, 1);
        __builtin_prefetch(Vg + ((size_t)(kv0 + KT + kr_) * NH + h) * DH + kcb, 0, 1);
      } else if (phase == 0) {
        __builtin_prefetch(k_remote + ((size_t)(ks + kr_) * NH + h) * DH + kcb, 0, 1);
        __builtin_prefetch(v_remote + ((size_t)(ks + kr_) * NH + h) * DH + kcb, 0, 1);
      }
      __syncthreads();

      if (!(causal && kv0 > q0 + WQT - 1)) {
        // ---- S = Q*K^T : two 16x16 n-tiles, K-depth 128 as 4x32; batch DS loads before WMMAs ----
        v8f s0 = vzero, s1 = vzero;
        {
          FragB bk[4];
          #pragma unroll
          for (int kc = 0; kc < 4; ++kc) {
            const int b0 = ln16 * KPAD + kc * 32 + hf * 8;        // kv col = ln16
            bk[kc].q[0] = *(const uint4*)&Klds[b0];
            bk[kc].q[1] = *(const uint4*)&Klds[b0 + 16];
          }
          #pragma unroll
          for (int kc = 0; kc < 4; ++kc)
            s0 = __builtin_amdgcn_wmma_f32_16x16x32_bf16(false, aQ[kc].v, false, bk[kc].v, (short)0, s0, false, false);
          #pragma unroll
          for (int kc = 0; kc < 4; ++kc) {
            const int b1 = (16 + ln16) * KPAD + kc * 32 + hf * 8; // kv col = 16+ln16
            bk[kc].q[0] = *(const uint4*)&Klds[b1];
            bk[kc].q[1] = *(const uint4*)&Klds[b1 + 16];
          }
          #pragma unroll
          for (int kc = 0; kc < 4; ++kc)
            s1 = __builtin_amdgcn_wmma_f32_16x16x32_bf16(false, aQ[kc].v, false, bk[kc].v, (short)0, s1, false, false);
        }

        // ---- online softmax; row(g) = q0 + g + 8*hf, cols across the 16-lane half ----
        const int qlo = q0 + 8 * hf;
        float sa[8], sb[8];
        #pragma unroll
        for (int g = 0; g < 8; ++g) {
          float x0 = s0[g] * scale + bias;
          float x1 = s1[g] * scale + bias;
          if (causal) {
            const int qg = qlo + g;
            if (kv0 + ln16 > qg)      x0 = -__builtin_inff();
            if (kv0 + 16 + ln16 > qg) x1 = -__builtin_inff();
          }
          sa[g] = x0; sb[g] = x1;
        }
        float corr[8];
        #pragma unroll
        for (int g = 0; g < 8; ++g) {
          const float rmax = redmax16(fmaxf(sa[g], sb[g]));
          const float mn   = fmaxf(m[g], rmax);
          const float c    = __expf(m[g] - mn);
          const float p0   = __expf(sa[g] - mn);
          const float p1   = __expf(sb[g] - mn);
          const float rs   = redsum16(p0 + p1);
          l[g] = l[g] * c + rs;
          m[g] = mn;
          corr[g] = c;
          Plds[wave][(g + 8 * hf) * PPAD + ln16]      = f2bf(p0);
          Plds[wave][(g + 8 * hf) * PPAD + 16 + ln16] = f2bf(p1);
        }
        #pragma unroll
        for (int t = 0; t < 8; ++t) {
          #pragma unroll
          for (int g = 0; g < 8; ++g) o[t][g] *= corr[g];
        }

        // ---- O += P*V : P re-laid-out via LDS into A-fragment form ----
        FragB a;
        const int pb = ln16 * PPAD + hf * 8;
        a.q[0] = *(const uint4*)&Plds[wave][pb];
        a.q[1] = *(const uint4*)&Plds[wave][pb + 16];
        #pragma unroll
        for (int tt = 0; tt < 8; tt += 4) {
          FragB bv[4];
          #pragma unroll
          for (int u = 0; u < 4; ++u) {
            const int vb = ((tt + u) * 16 + ln16) * VPAD + hf * 8; // d col = (tt+u)*16+ln16
            bv[u].q[0] = *(const uint4*)&Vtld[vb];
            bv[u].q[1] = *(const uint4*)&Vtld[vb + 16];
          }
          #pragma unroll
          for (int u = 0; u < 4; ++u)
            o[tt + u] = __builtin_amdgcn_wmma_f32_16x16x32_bf16(false, a.v, false, bv[u].v, (short)0, o[tt + u], false, false);
        }
      }
      __syncthreads();
    }
  }

  // ---- epilogue: out = O/l, lse = m + log l ----
  #pragma unroll
  for (int g = 0; g < 8; ++g) {
    const int r = q0 + g + 8 * hf;
    const float inv = 1.0f / l[g];
    float* orow = out + ((size_t)r * NH + h) * DH;
    #pragma unroll
    for (int t = 0; t < 8; ++t) orow[t * 16 + ln16] = o[t][g] * inv;
    if (ln16 == g) lse[(size_t)h * SQ + r] = m[g] + __logf(l[g]);
  }
}

extern "C" void kernel_launch(void* const* d_in, const int* in_sizes, int n_in,
                              void* d_out, int out_size, void* d_ws, size_t ws_size,
                              hipStream_t stream) {
  const float* q        = (const float*)d_in[0];
  const float* k_local  = (const float*)d_in[1];
  const float* v_local  = (const float*)d_in[2];
  const float* k_remote = (const float*)d_in[3];
  const float* v_remote = (const float*)d_in[4];
  float* out = (float*)d_out;
  float* lse = out + (size_t)SQ * NH * DH;
  dim3 grid(SQ / QT, NH);
  dist_flash_attn_kernel<<<grid, 128, 0, stream>>>(q, k_local, v_local,
                                                   k_remote, v_remote, out, lse);
}
